// OpenPanguAttention_71030169141400
// MI455X (gfx1250) — compile-verified, matrix operand order checked by reading, not run on hardware
//
#include <hip/hip_runtime.h>
#include <hip/hip_bf16.h>

// ---------------------------------------------------------------------------
// OpenPangu attention for MI455X (gfx1250, wave32, WMMA bf16 path)
//   T=2048 HIDDEN=4096 NH=32 NKV=8 HD=128 QKV_OUT=6144
// Pipeline: qkv GEMM (bf16 wmma) -> rope -> flash attention (bf16 wmma,
//           async K staging via GLOBAL_LOAD_ASYNC_TO_LDS) -> out GEMM
// ---------------------------------------------------------------------------

typedef __bf16 v16bf __attribute__((ext_vector_type(16)));
typedef float  v8f   __attribute__((ext_vector_type(8)));

union bfrag_u { v16bf v; unsigned u[8]; };

__device__ __forceinline__ unsigned short f2bf(float f) {
    union { float f; unsigned u; } v; v.f = f;
    unsigned r = v.u + 0x7fffu + ((v.u >> 16) & 1u);
    return (unsigned short)(r >> 16);
}
__device__ __forceinline__ float bf2f(unsigned short h) {
    union { unsigned u; float f; } v; v.u = ((unsigned)h) << 16;
    return v.f;
}

// K-pair base for 16-bit A/B fragment VGPR v (per CDNA5 ISA 7.12.2):
// lanes 0-15:  v0..3 -> K{0,1},{2,3},{4,5},{6,7}; v4..7 -> K{16..23} pairs
// lanes 16-31: +8
__device__ __forceinline__ int kpair(int v, int half) {
    return (v < 4 ? 2 * v : 16 + 2 * (v - 4)) + 8 * half;
}

__device__ __forceinline__ float halfmax16(float v) {
    #pragma unroll
    for (int m = 1; m < 16; m <<= 1) v = fmaxf(v, __shfl_xor(v, m, 32));
    return v;
}
__device__ __forceinline__ float halfsum16(float v) {
    #pragma unroll
    for (int m = 1; m < 16; m <<= 1) v += __shfl_xor(v, m, 32);
    return v;
}

// ---- CDNA5 async global->LDS copy (ASYNCcnt), 16B per lane ----------------
__device__ __forceinline__ void async_copy_b128(unsigned lds_off, const void* gptr) {
    asm volatile("global_load_async_to_lds_b128 %0, %1, off"
                 :: "v"(lds_off), "v"((unsigned long long)gptr)
                 : "memory");
}
__device__ __forceinline__ void wait_async_all() {
    asm volatile("s_wait_asynccnt 0" ::: "memory");
}

// ---------------------------------------------------------------------------
// Generic tiled bf16-WMMA GEMM: C[M,N] = A[M,K] * B
//   B_TRANS: B[k][n] = W[n*K + k]   (w_qkv: out = hidden @ W^T)
//   else:    B[k][n] = W[k*N + n]   (w_o)
// Block: 256 threads (8 waves), tile 64(M) x 128(N), BK=32.
// Wave grid 2(M) x 4(N); each wave -> 32x32 = 2x2 wmma tiles.
// ---------------------------------------------------------------------------
template<bool A_BF16, bool B_TRANS, bool OUT_BF16>
__global__ __launch_bounds__(256)
void gemm_wmma(const void* __restrict__ Aptr, const void* __restrict__ Bptr,
               void* __restrict__ Cptr, int M, int N, int K)
{
    __shared__ unsigned short As[64 * 34];   // [m][k] pad->34
    __shared__ unsigned short Bs[128 * 34];  // [n][k] pad->34

    const int tid   = threadIdx.x;
    const int lane  = tid & 31;
    const int wid   = tid >> 5;
    const int waveM = wid >> 2;      // 0..1
    const int waveN = wid & 3;       // 0..3
    const int half  = lane >> 4;
    const int r16   = lane & 15;
    const int rowBase = blockIdx.y * 64;
    const int colBase = blockIdx.x * 128;

    v8f acc[2][2] = {};

    for (int k0 = 0; k0 < K; k0 += 32) {
        __syncthreads();
        // ---- stage A tile (64 x 32) ----
        if (A_BF16) {
            const unsigned* Ag = (const unsigned*)Aptr;  // bf16 pairs
            #pragma unroll
            for (int i = 0; i < 4; ++i) {
                int u = tid + i * 256;          // 1024 uints
                int r = u >> 4, c2 = u & 15;
                unsigned val = Ag[(size_t)(rowBase + r) * (K >> 1) + (k0 >> 1) + c2];
                *(unsigned*)&As[r * 34 + 2 * c2] = val;
            }
        } else {
            const float* Ag = (const float*)Aptr;
            #pragma unroll
            for (int i = 0; i < 8; ++i) {
                int u = tid + i * 256;          // 2048 floats
                int r = u >> 5, c = u & 31;
                As[r * 34 + c] = f2bf(Ag[(size_t)(rowBase + r) * K + k0 + c]);
            }
        }
        // ---- stage B tile (128 n x 32 k), normalized to [n][k] ----
        {
            const float* Bg = (const float*)Bptr;
            if (B_TRANS) {
                #pragma unroll
                for (int i = 0; i < 16; ++i) {
                    int u = tid + i * 256;      // 4096 floats
                    int n = u >> 5, c = u & 31;
                    Bs[n * 34 + c] = f2bf(Bg[(size_t)(colBase + n) * K + k0 + c]);
                }
            } else {
                #pragma unroll
                for (int i = 0; i < 16; ++i) {
                    int u = tid + i * 256;
                    int k = u >> 7, n = u & 127;
                    Bs[n * 34 + k] = f2bf(Bg[(size_t)(k0 + k) * N + colBase + n]);
                }
            }
        }
        // ---- warm L2 for the next K tile while we compute this one ----
        if (k0 + 32 < K) {
            int kn = k0 + 32;
            if (A_BF16) {
                const char* pa = (const char*)Aptr +
                    ((size_t)(rowBase + (tid >> 3)) * K + kn) * 2;
                __builtin_prefetch(pa, 0, 1);
            } else {
                const char* pa = (const char*)Aptr +
                    ((size_t)(rowBase + (tid >> 2)) * K + kn) * 4;
                __builtin_prefetch(pa, 0, 1);
            }
            if (B_TRANS) {
                const char* pb = (const char*)Bptr +
                    ((size_t)(colBase + (tid >> 1)) * K + kn) * 4;
                __builtin_prefetch(pb, 0, 1);
            } else {
                const char* pb = (const char*)Bptr +
                    ((size_t)(kn + (tid >> 3)) * N + colBase + (tid & 7) * 16) * 4;
                __builtin_prefetch(pb, 0, 1);
            }
        }
        __syncthreads();

        // ---- build fragments + 2x2 WMMA ----
        bfrag_u af[2], bfr[2];
        #pragma unroll
        for (int t = 0; t < 2; ++t) {
            int ar = waveM * 32 + t * 16 + r16;
            int bn = waveN * 32 + t * 16 + r16;
            #pragma unroll
            for (int v = 0; v < 8; ++v) {
                int kp = kpair(v, half);
                af[t].u[v]  = *(const unsigned*)&As[ar * 34 + kp];
                bfr[t].u[v] = *(const unsigned*)&Bs[bn * 34 + kp];
            }
        }
        #pragma unroll
        for (int tm = 0; tm < 2; ++tm)
            #pragma unroll
            for (int tn = 0; tn < 2; ++tn)
                acc[tm][tn] = __builtin_amdgcn_wmma_f32_16x16x32_bf16(
                    false, af[tm].v, false, bfr[tn].v,
                    (short)0, acc[tm][tn], false, false);
    }

    // ---- epilogue (C/D layout: VGPR g -> M = g + 8*half, N = lane&15) ----
    #pragma unroll
    for (int tm = 0; tm < 2; ++tm)
        #pragma unroll
        for (int tn = 0; tn < 2; ++tn)
            #pragma unroll
            for (int g = 0; g < 8; ++g) {
                int r = rowBase + waveM * 32 + tm * 16 + g + 8 * half;
                int c = colBase + waveN * 32 + tn * 16 + r16;
                float val = acc[tm][tn][g];
                if (OUT_BF16)
                    ((unsigned short*)Cptr)[(size_t)r * N + c] = f2bf(val);
                else
                    ((float*)Cptr)[(size_t)r * N + c] = val;
            }
}

// ---------------------------------------------------------------------------
// NEOX RoPE on Q (heads 0..31) and K (heads 32..39) of the bf16 qkv buffer.
// grid (T, 40), block 64: thread d handles pair (d, d+64) in one head.
// ---------------------------------------------------------------------------
__global__ __launch_bounds__(64)
void rope_kernel(unsigned short* __restrict__ qkv, const int* __restrict__ positions)
{
    const int t = blockIdx.x;
    const int h = blockIdx.y;                 // 0..39
    const int d = threadIdx.x;                // 0..63
    const int base = (h < 32) ? h * 128 : 4096 + (h - 32) * 128;

    float pos  = (float)positions[t];
    float inv  = powf(10000.0f, -(float)d * (1.0f / 64.0f));
    float fr   = pos * inv;
    float c    = cosf(fr), s = sinf(fr);

    size_t o = (size_t)t * 6144 + base + d;
    float x1 = bf2f(qkv[o]);
    float x2 = bf2f(qkv[o + 64]);
    qkv[o]      = f2bf(x1 * c - x2 * s);
    qkv[o + 64] = f2bf(x2 * c + x1 * s);
}

// ---------------------------------------------------------------------------
// Flash attention: grid (32 heads, 32 M-blocks of 64 rows), block 128 (4 waves)
// Each wave owns a 16-row Q strip -> softmax stats reduce in a half-wave.
// K tile staged via GLOBAL_LOAD_ASYNC_TO_LDS_B128 (ASYNCcnt); V staged with
// an in-flight transpose through VGPRs (needed for P@V B-fragment k-pairs).
// ---------------------------------------------------------------------------
__global__ __launch_bounds__(128)
void flash_kernel(const unsigned short* __restrict__ qkv,
                  unsigned short* __restrict__ attn)
{
    __shared__ unsigned short Ks[64 * 136];    // [kv_row][d]   (pad 136)
    __shared__ unsigned short Vst[128 * 72];   // [d][kv_row]   (pad 72) transposed
    __shared__ unsigned short Ps[4][16 * 72];  // per-wave P staging [row][kv]

    const int head = blockIdx.x;               // 0..31
    const int mb   = blockIdx.y;               // 0..31
    const int kvh  = head >> 2;                // GQA: 4 q-heads per kv-head
    const int tid  = threadIdx.x;
    const int lane = tid & 31;
    const int wid  = tid >> 5;
    const int half = lane >> 4;
    const int r16  = lane & 15;
    const int qRow0 = mb * 64 + wid * 16;

    const unsigned* qg = (const unsigned*)qkv;      // bf16 pairs, row stride 3072
    const int vColU = (5120 + kvh * 128) >> 1;      // uint offset of V section
    const unsigned ksOff = (unsigned)(unsigned long long)(void*)&Ks[0];

    // persistent Q fragments: 16 rows x 128 dims = 4 fragments of 16x32
    bfrag_u qf[4];
    #pragma unroll
    for (int kc = 0; kc < 4; ++kc)
        #pragma unroll
        for (int v = 0; v < 8; ++v) {
            int kp  = kpair(v, half);
            int col = head * 128 + kc * 32 + kp;
            qf[kc].u[v] = qg[(size_t)(qRow0 + r16) * 3072 + (col >> 1)];
        }

    v8f oacc[8] = {};
    float m_i[8], l_i[8];
    #pragma unroll
    for (int g = 0; g < 8; ++g) { m_i[g] = -3.0e38f; l_i[g] = 0.0f; }

    const float SCALE = 0.08838834764831845f;   // 1/sqrt(128)
    const int nkb = mb + 1;                     // causal: kv blocks 0..mb

    for (int kb = 0; kb < nkb; ++kb) {
        __syncthreads();   // previous iteration done reading Ks/Vst

        // ---- K tile: pure byte copy, 64 rows x 256B -> async direct to LDS
        {
            const char* kbase = (const char*)qkv +
                ((size_t)(kb * 64) * 6144 + 4096 + kvh * 128) * 2;
            #pragma unroll
            for (int i = 0; i < 8; ++i) {
                int idx = tid + i * 128;            // 1024 x 16B segments
                int r = idx >> 4, seg = idx & 15;
                async_copy_b128(ksOff + (unsigned)(r * 272 + seg * 16),
                                kbase + (size_t)r * 12288 + seg * 16);
            }
        }
        // ---- V tile: transpose in flight -> Vst[d][kv] ----
        #pragma unroll
        for (int i = 0; i < 32; ++i) {
            int u = tid + i * 128;              // 4096 uints
            int r = u >> 6, d2 = u & 63;
            unsigned vval = qg[(size_t)(kb * 64 + r) * 3072 + vColU + d2];
            Vst[(2 * d2)     * 72 + r] = (unsigned short)(vval & 0xffffu);
            Vst[(2 * d2 + 1) * 72 + r] = (unsigned short)(vval >> 16);
        }
        wait_async_all();
        __syncthreads();

        // ---- S = Q @ K^T : 4 column tiles of 16, K-dim 128 in 4 chunks ----
        v8f sacc[4] = {};
        #pragma unroll
        for (int ct = 0; ct < 4; ++ct) {
            #pragma unroll
            for (int kc = 0; kc < 4; ++kc) {
                bfrag_u kf;
                #pragma unroll
                for (int v = 0; v < 8; ++v) {
                    int kp = kpair(v, half);
                    kf.u[v] = *(const unsigned*)&Ks[(ct * 16 + r16) * 136 + kc * 32 + kp];
                }
                sacc[ct] = __builtin_amdgcn_wmma_f32_16x16x32_bf16(
                    false, qf[kc].v, false, kf.v, (short)0, sacc[ct], false, false);
            }
        }

        // ---- online softmax (rows owned within a 16-lane half) ----
        const int rowG0 = qRow0 + 8 * half;
        #pragma unroll
        for (int g = 0; g < 8; ++g) {
            int rg = rowG0 + g;
            float mx = -3.0e38f;
            #pragma unroll
            for (int ct = 0; ct < 4; ++ct) {
                float s = sacc[ct][g] * SCALE;
                int cg = kb * 64 + ct * 16 + r16;
                if (cg > rg) s = -3.0e38f;      // causal mask
                sacc[ct][g] = s;
                mx = fmaxf(mx, s);
            }
            mx = halfmax16(mx);
            float mnew  = fmaxf(m_i[g], mx);
            float alpha = __expf(m_i[g] - mnew);
            float rs = 0.0f;
            #pragma unroll
            for (int ct = 0; ct < 4; ++ct) {
                float p = __expf(sacc[ct][g] - mnew);
                rs += p;
                Ps[wid][(g + 8 * half) * 72 + ct * 16 + r16] = f2bf(p);
            }
            rs = halfsum16(rs);
            l_i[g] = l_i[g] * alpha + rs;
            m_i[g] = mnew;
            #pragma unroll
            for (int j = 0; j < 8; ++j) oacc[j][g] *= alpha;
        }
        __syncthreads();   // P visible in LDS (uniform across waves)

        // ---- O += P @ V : 8 dim tiles of 16, kv-dim 64 in 4 chunks ----
        bfrag_u pf[4];
        #pragma unroll
        for (int kc = 0; kc < 4; ++kc)
            #pragma unroll
            for (int v = 0; v < 8; ++v) {
                int kp = kpair(v, half);
                pf[kc].u[v] = *(const unsigned*)&Ps[wid][r16 * 72 + kc * 32 + kp];
            }
        #pragma unroll
        for (int j = 0; j < 8; ++j) {
            #pragma unroll
            for (int kc = 0; kc < 4; ++kc) {
                bfrag_u vf;
                #pragma unroll
                for (int v = 0; v < 8; ++v) {
                    int kp = kpair(v, half);
                    vf.u[v] = *(const unsigned*)&Vst[(j * 16 + r16) * 72 + kc * 32 + kp];
                }
                oacc[j] = __builtin_amdgcn_wmma_f32_16x16x32_bf16(
                    false, pf[kc].v, false, vf.v, (short)0, oacc[j], false, false);
            }
        }
    }

    // ---- write attn output (bf16) ----
    #pragma unroll
    for (int j = 0; j < 8; ++j)
        #pragma unroll
        for (int g = 0; g < 8; ++g) {
            int r = qRow0 + g + 8 * half;
            int c = head * 128 + j * 16 + r16;
            attn[(size_t)r * 4096 + c] = f2bf(oacc[j][g] / l_i[g]);
        }
}

// ---------------------------------------------------------------------------
extern "C" void kernel_launch(void* const* d_in, const int* in_sizes, int n_in,
                              void* d_out, int out_size, void* d_ws, size_t ws_size,
                              hipStream_t stream)
{
    (void)in_sizes; (void)n_in; (void)out_size; (void)ws_size;

    const int*   positions = (const int*)d_in[0];
    const float* hidden    = (const float*)d_in[1];   // [2048, 4096]
    const float* w_qkv     = (const float*)d_in[2];   // [6144, 4096]
    const float* w_o       = (const float*)d_in[3];   // [4096, 4096]
    float*       out       = (float*)d_out;           // [2048, 4096]

    unsigned short* qkv  = (unsigned short*)d_ws;                          // 2048*6144 bf16 (24 MB)
    unsigned short* attn = (unsigned short*)((char*)d_ws +
                                             (size_t)2048 * 6144 * 2);     // 2048*4096 bf16 (16 MB)

    // 1) qkv = hidden @ w_qkv^T  (fp32 in, bf16 out)
    gemm_wmma<false, true, true>
        <<<dim3(6144 / 128, 2048 / 64), dim3(256), 0, stream>>>(
            hidden, w_qkv, qkv, 2048, 6144, 4096);

    // 2) RoPE on Q + K in place
    rope_kernel<<<dim3(2048, 40), dim3(64), 0, stream>>>(qkv, positions);

    // 3) causal GQA flash attention -> attn (bf16)
    flash_kernel<<<dim3(32, 32), dim3(128), 0, stream>>>(qkv, attn);

    // 4) out = attn @ w_o  (bf16 A, fp32 B, fp32 out)
    gemm_wmma<true, false, false>
        <<<dim3(4096 / 128, 2048 / 64), dim3(256), 0, stream>>>(
            attn, w_o, out, 2048, 4096, 4096);
}
